// SpliceMamba_75479755260149
// MI455X (gfx1250) — matrix operand order repaired and checked by reading, not compiled, not run on hardware
//
#include <hip/hip_runtime.h>
#include <hip/hip_bf16.h>
#include <math.h>

// ---------------- model constants ----------------
#define B_      2
#define L_      2048
#define M_      (B_*L_)        // 4096 tokens
#define DMODEL  256
#define DINNER  256
#define DSTATE  16
#define DTRANK  16
#define NHEADS  8
#define HDIM    32
#define RADIUS_ 200
#define EPSF    1e-5f

typedef __attribute__((ext_vector_type(16))) _Float16 v16h;
typedef __attribute__((ext_vector_type(8)))  _Float16 v8h;
typedef __attribute__((ext_vector_type(8)))  float    v8f;

// ---------------- device helpers ----------------
__device__ __forceinline__ float gelu_f(float x) {
    return 0.5f * x * (1.0f + erff(x * 0.70710678118654752f));
}
__device__ __forceinline__ float silu_f(float x) {
    return x / (1.0f + __expf(-x));
}
__device__ __forceinline__ float softplus_f(float x) {
    return (x > 20.f) ? x : log1pf(__expf(x));
}
__device__ __forceinline__ float wave_sum(float v) {
    #pragma unroll
    for (int off = 16; off > 0; off >>= 1) v += __shfl_xor(v, off, 32);
    return v;
}

// ---------------- f32 -> f16 convert ----------------
__global__ void cvt_kernel(const float* __restrict__ s, _Float16* __restrict__ d, int n) {
    int i = blockIdx.x * blockDim.x + threadIdx.x;
    if (i < n) d[i] = (_Float16)s[i];
}

// ---------------- WMMA GEMM: C[M,N] = act(A[M,K] @ W[N,K]^T + bias) (+res) ----------------
// A, W f16 row-major; K multiple of 32; M multiple of 64.
// Each wave owns a 4x2 grid of 16x16 tiles (64 rows x 32 cols): per K-step the two
// B fragments are reused across 4 A fragments -> 8 WMMAs per 12 b128 loads.
#define MT 4
#define NT 2

__global__ __launch_bounds__(128)
void gemm_wmma_kernel(const _Float16* __restrict__ A,
                      const _Float16* __restrict__ W,
                      const float* __restrict__ bias,
                      const float* __restrict__ res,
                      float* __restrict__ C,
                      _Float16* __restrict__ C16,
                      int M, int N, int K, int act)
{
    const int wave   = threadIdx.x >> 5;
    const int lane   = threadIdx.x & 31;
    const int tilesN = N >> 4;
    const int ngrp   = (tilesN + NT - 1) / NT;
    const int mgrp   = (M >> 4) / MT;
    const int job    = blockIdx.x * (blockDim.x >> 5) + wave;
    if (job >= mgrp * ngrp) return;                 // wave-uniform: EXEC all-ones
    const int jm = job / ngrp;
    const int jn = job - jm * ngrp;
    const int half = lane >> 4;
    const int r    = lane & 15;

    const int tn0 = jn * NT;
    int tn1 = tn0 + 1;
    const bool n1ok = (tn1 < tilesN);               // wave-uniform
    if (!n1ok) tn1 = tn0;                           // clamp: compute dup, skip store

    // 16-bit operand k order used by this code (contiguous; A and B agree):
    //  A 16x32: lane(half,m): VGPR0..7 = K half*16+0..15 (same k order as B)
    //  B 32x16: lane(half,n): VGPR0..7 = K half*16+0..15 (contiguous row of W)
    const _Float16* ar[MT];
    #pragma unroll
    for (int s = 0; s < MT; s++)
        ar[s] = A + (size_t)((jm * MT + s) * 16 + r) * K + half * 16;
    const _Float16* wr0 = W + (size_t)(tn0 * 16 + r) * K + half * 16;
    const _Float16* wr1 = W + (size_t)(tn1 * 16 + r) * K + half * 16;

    v8f acc[MT][NT] = {};
    for (int k0 = 0; k0 < K; k0 += 32) {
        __builtin_prefetch(wr0 + k0 + 32, 0, 3);    // WGP-scope bias; spec. prefetch
        __builtin_prefetch(ar[0] + k0 + 32, 0, 3);  // OOB speculative is dropped
        v16h b0 = *(const v16h*)(wr0 + k0);
        v16h b1 = *(const v16h*)(wr1 + k0);
        #pragma unroll
        for (int s = 0; s < MT; s++) {
            v8h alo = *(const v8h*)(ar[s] + k0);
            v8h ahi = *(const v8h*)(ar[s] + k0 + 8);
            v16h af;
            #pragma unroll
            for (int i = 0; i < 8; i++) { af[i] = alo[i]; af[8 + i] = ahi[i]; }
            acc[s][0] = __builtin_amdgcn_wmma_f32_16x16x32_f16(
                            false, af, false, b0, (short)0, acc[s][0], false, false);
            acc[s][1] = __builtin_amdgcn_wmma_f32_16x16x32_f16(
                            false, af, false, b1, (short)0, acc[s][1], false, false);
        }
    }

    #pragma unroll
    for (int nt = 0; nt < NT; nt++) {
        if (nt == 1 && !n1ok) break;                // wave-uniform
        const int col = ((nt ? tn1 : tn0) << 4) + r;   // D layout: N in lane&15
        const float bv = bias ? bias[col] : 0.f;
        #pragma unroll
        for (int s = 0; s < MT; s++) {
            #pragma unroll
            for (int j = 0; j < 8; j++) {
                const int row = (jm * MT + s) * 16 + j + (half << 3); // M = vgpr+8*half
                float vv = acc[s][nt][j] + bv;
                if (act == 1) vv = gelu_f(vv);
                if (res) vv += res[(size_t)row * N + col];
                C[(size_t)row * N + col] = vv;
                if (C16) C16[(size_t)row * N + col] = (_Float16)vv;
            }
        }
    }
}

// ---------------- stem ----------------
// conv1: x[B,4,L] (k=11, pad 5) -> bn -> gelu -> h1[B,L,128]
__global__ void stem1_kernel(const float* __restrict__ x, const float* __restrict__ w1,
                             const float* __restrict__ b1, const float* __restrict__ g,
                             const float* __restrict__ bb, const float* __restrict__ mn,
                             const float* __restrict__ vr, float* __restrict__ h1)
{
    int idx = blockIdx.x * blockDim.x + threadIdx.x;      // B*L*128
    if (idx >= M_ * 128) return;
    int c = idx & 127;
    int t = (idx >> 7) % L_;
    int b = idx / (128 * L_);
    float acc = b1[c];
    #pragma unroll
    for (int ic = 0; ic < 4; ic++)
        #pragma unroll
        for (int k = 0; k < 11; k++) {
            int l = t + k - 5;
            if (l >= 0 && l < L_)
                acc += x[((size_t)b * 4 + ic) * L_ + l] * w1[(c * 4 + ic) * 11 + k];
        }
    acc = (acc - mn[c]) * rsqrtf(vr[c] + EPSF) * g[c] + bb[c];
    h1[((size_t)b * L_ + t) * 128 + c] = gelu_f(acc);
}

// conv2 (1x1, 128->256) -> bn -> gelu -> +posenc ; writes hf and reversed hb
__global__ void stem2_kernel(const float* __restrict__ h1, const float* __restrict__ w2,
                             const float* __restrict__ b2, const float* __restrict__ g,
                             const float* __restrict__ bb, const float* __restrict__ mn,
                             const float* __restrict__ vr,
                             float* __restrict__ hf, float* __restrict__ hbrev)
{
    int idx = blockIdx.x * blockDim.x + threadIdx.x;      // B*L*256
    if (idx >= M_ * DMODEL) return;
    int c = idx & 255;
    int t = (idx >> 8) % L_;
    int b = idx / (DMODEL * L_);
    const float* hrow = h1 + ((size_t)b * L_ + t) * 128;
    float acc = b2[c];
    #pragma unroll 8
    for (int ci = 0; ci < 128; ci++) acc += hrow[ci] * w2[c * 128 + ci];
    acc = (acc - mn[c]) * rsqrtf(vr[c] + EPSF) * g[c] + bb[c];
    acc = gelu_f(acc);
    float freq = __expf(-(float)(c & ~1) * (9.210340371976184f / 256.f)); // ln(1e4)/d
    float ang  = (float)t * freq;
    acc += (c & 1) ? cosf(ang) : sinf(ang);
    hf[((size_t)b * L_ + t) * DMODEL + c] = acc;
    hbrev[((size_t)b * L_ + (L_ - 1 - t)) * DMODEL + c] = acc;
}

// ---------------- layernorm (wave per token), optional gated mix in front ----------------
__global__ void ln_kernel(const float* __restrict__ x, const float* __restrict__ op,
                          const float* __restrict__ gate, const float* __restrict__ g,
                          const float* __restrict__ b,
                          float* __restrict__ out32, _Float16* __restrict__ out16)
{
    int wid  = (blockIdx.x * blockDim.x + threadIdx.x) >> 5;   // token
    int lane = threadIdx.x & 31;
    if (wid >= M_) return;
    const float* row = x + (size_t)wid * DMODEL;
    float gv = gate ? gate[wid] : 0.f;
    float v[8];
    #pragma unroll
    for (int i = 0; i < 8; i++) {
        int c = lane + 32 * i;
        float a = row[c];
        if (op) a = a * (1.f - gv) + op[(size_t)wid * DMODEL + c] * gv;
        v[i] = a;
    }
    float s = 0.f;
    #pragma unroll
    for (int i = 0; i < 8; i++) s += v[i];
    s = wave_sum(s);
    float mean = s * (1.f / 256.f);
    float q = 0.f;
    #pragma unroll
    for (int i = 0; i < 8; i++) { float d = v[i] - mean; q += d * d; }
    q = wave_sum(q);
    float inv = rsqrtf(q * (1.f / 256.f) + EPSF);
    #pragma unroll
    for (int i = 0; i < 8; i++) {
        int c = lane + 32 * i;
        float yv = (v[i] - mean) * inv * g[c] + b[c];
        if (out32) out32[(size_t)wid * DMODEL + c] = yv;
        if (out16) out16[(size_t)wid * DMODEL + c] = (_Float16)yv;
    }
}

// ---------------- depthwise causal conv (k=4) + silu on xi half of xz ----------------
__global__ void dwconv_silu_kernel(const float* __restrict__ xz, const float* __restrict__ cw,
                                   const float* __restrict__ cb,
                                   float* __restrict__ xia, _Float16* __restrict__ xia16)
{
    int idx = blockIdx.x * blockDim.x + threadIdx.x;      // B*L*256, [b][t][d]
    if (idx >= M_ * DINNER) return;
    int d = idx & 255;
    int t = (idx >> 8) % L_;
    int b = idx / (DINNER * L_);
    float acc = cb[d];
    #pragma unroll
    for (int k = 0; k < 4; k++) {
        int l = t - 3 + k;
        if (l >= 0) acc += xz[((size_t)b * L_ + l) * 512 + d] * cw[d * 4 + k];
    }
    float v = silu_f(acc);
    xia[idx]   = v;
    xia16[idx] = (_Float16)v;
}

// ---------------- dt = softplus(dbc[:, :16] @ dtW^T + dtB) ----------------
__global__ void dt_kernel(const float* __restrict__ dbc, const float* __restrict__ dtw,
                          const float* __restrict__ dtb, float* __restrict__ dt)
{
    int idx = blockIdx.x * blockDim.x + threadIdx.x;      // M*256
    if (idx >= M_ * DINNER) return;
    int d = idx & 255;
    int m = idx >> 8;
    const float* row = dbc + (size_t)m * 48;
    float acc = dtb[d];
    #pragma unroll
    for (int r = 0; r < DTRANK; r++) acc += row[r] * dtw[d * DTRANK + r];
    dt[idx] = softplus_f(acc);
}

// ---------------- sequential selective-scan; one thread per (b,d), 16 states in regs ----------------
__global__ void ssm_scan_kernel(const float* __restrict__ dt, const float* __restrict__ dbc,
                                const float* __restrict__ xia, const float* __restrict__ xz,
                                const float* __restrict__ A_log, const float* __restrict__ Dp,
                                _Float16* __restrict__ y16)
{
    int b = blockIdx.x;
    int d = threadIdx.x;                 // 256 channels
    float Ac[DSTATE], hst[DSTATE];
    #pragma unroll
    for (int s = 0; s < DSTATE; s++) { Ac[s] = -__expf(A_log[d * DSTATE + s]); hst[s] = 0.f; }
    const float Dd = Dp[d];
    for (int t = 0; t < L_; t++) {
        size_t tok = (size_t)b * L_ + t;
        float dtv = dt[tok * DINNER + d];
        float u   = xia[tok * DINNER + d];
        const float* bc = dbc + tok * 48;      // [dt(16) | B(16) | C(16)]
        float acc = 0.f;
        #pragma unroll
        for (int s = 0; s < DSTATE; s++) {
            float dA = __expf(dtv * Ac[s]);
            hst[s] = dA * hst[s] + dtv * bc[16 + s] * u;
            acc += hst[s] * bc[32 + s];
        }
        float yv = acc + Dd * u;
        float z  = xz[tok * 512 + 256 + d];
        yv *= silu_f(z);
        y16[tok * DINNER + d] = (_Float16)yv;
    }
}

// ---------------- reverse along L ----------------
__global__ void reverse_kernel(const float* __restrict__ in, float* __restrict__ out)
{
    int idx = blockIdx.x * blockDim.x + threadIdx.x;      // M*256
    if (idx >= M_ * DMODEL) return;
    int c = idx & 255;
    int t = (idx >> 8) % L_;
    int b = idx / (DMODEL * L_);
    out[idx] = in[((size_t)b * L_ + (L_ - 1 - t)) * DMODEL + c];
}

// ---------------- [hf | hb] -> f16 concat ----------------
__global__ void concat16_kernel(const float* __restrict__ hf, const float* __restrict__ hb,
                                _Float16* __restrict__ cat)
{
    int idx = blockIdx.x * blockDim.x + threadIdx.x;      // M*512
    if (idx >= M_ * 512) return;
    int c = idx & 511;
    int m = idx >> 9;
    float v = (c < 256) ? hf[(size_t)m * 256 + c] : hb[(size_t)m * 256 + (c - 256)];
    cat[idx] = (_Float16)v;
}

// ---------------- head second matmul (N=3) + optional gate ----------------
__global__ void head2_kernel(const float* __restrict__ hh, const float* __restrict__ w2,
                             const float* __restrict__ b2, float* __restrict__ out,
                             float* __restrict__ gate)
{
    int m = blockIdx.x * blockDim.x + threadIdx.x;
    if (m >= M_) return;
    const float* row = hh + (size_t)m * DMODEL;
    float o[3];
    #pragma unroll
    for (int c = 0; c < 3; c++) {
        float acc = b2[c];
        #pragma unroll 8
        for (int k = 0; k < DMODEL; k++) acc += row[k] * w2[c * DMODEL + k];
        o[c] = acc;
        out[(size_t)m * 3 + c] = acc;
    }
    if (gate) gate[m] = 1.f / (1.f + __expf(-fmaxf(o[1], o[2])));
}

// ---------------- banded attention: wave per (b,h,l); lane = head-dim; online softmax ----------------
__global__ void band_attn_kernel(const float* __restrict__ q, const float* __restrict__ k,
                                 const float* __restrict__ v, _Float16* __restrict__ o16)
{
    int wid  = (blockIdx.x * blockDim.x + threadIdx.x) >> 5;  // b*H*L
    int lane = threadIdx.x & 31;
    if (wid >= B_ * NHEADS * L_) return;
    int l = wid % L_;
    int h = (wid / L_) & (NHEADS - 1);
    int b = wid / (NHEADS * L_);
    size_t base = (size_t)b * L_ * DMODEL + h * HDIM;
    float qd = q[base + (size_t)l * DMODEL + lane];
    int m0 = l - RADIUS_; if (m0 < 0) m0 = 0;
    int m1 = l + RADIUS_; if (m1 > L_ - 1) m1 = L_ - 1;
    float Mx = -3.4e38f, den = 0.f, oacc = 0.f;
    const float scale = 0.17677669529663687f;  // 1/sqrt(32)
    for (int m = m0; m <= m1; m++) {
        float kd = k[base + (size_t)m * DMODEL + lane];
        float p  = wave_sum(qd * kd) * scale;
        float nM = fmaxf(Mx, p);
        float corr = __expf(Mx - nM);
        float e    = __expf(p - nM);
        den  = den * corr + e;
        oacc = oacc * corr + e * v[base + (size_t)m * DMODEL + lane];
        Mx = nM;
    }
    o16[base + (size_t)l * DMODEL + lane] = (_Float16)(oacc / den);
}

// =======================================================================================
extern "C" void kernel_launch(void* const* d_in, const int* in_sizes, int n_in,
                              void* d_out, int out_size, void* d_ws, size_t ws_size,
                              hipStream_t stream)
{
    (void)in_sizes; (void)n_in; (void)out_size; (void)ws_size;
    const float* P[174];
    for (int i = 0; i < 174; i++) P[i] = (const float*)d_in[i];

    // param index map (setup_inputs dict order)
    auto LYR = [&](int dir, int i, int f) { return P[13 + dir * 66 + 11 * i + f]; };
    // fields: 0 ln_g 1 ln_b 2 in_proj_w 3 conv_w 4 conv_b 5 x_proj_w 6 dt_proj_w
    //         7 dt_proj_b 8 A_log 9 D 10 out_proj_w
    auto ATT = [&](int j, int f) { return P[150 + 10 * j + f]; };
    // fields: 0 qw 1 qb 2 kw 3 kb 4 vw 5 vb 6 ow 7 ob 8 ln_g 9 ln_b

    // ---------------- workspace layout ----------------
    size_t off = 0;
    auto alloc = [&](size_t bytes) -> void* {
        void* p = (char*)d_ws + off;
        off += (bytes + 255) & ~(size_t)255;
        return p;
    };
    auto aF16 = [&](size_t n) { return (_Float16*)alloc(n * sizeof(_Float16)); };
    auto aF32 = [&](size_t n) { return (float*)alloc(n * sizeof(float)); };

    _Float16 *inproj16[12], *xproj16[12], *outproj16[12];
    for (int i = 0; i < 12; i++) {
        inproj16[i]  = aF16(512 * 256);
        xproj16[i]   = aF16(48 * 256);
        outproj16[i] = aF16(256 * 256);
    }
    _Float16* fusion16 = aF16(256 * 512);
    _Float16* cw1_16   = aF16(256 * 256);
    _Float16* rw1_16   = aF16(256 * 256);
    _Float16* attw16[2][4];
    for (int j = 0; j < 2; j++)
        for (int f = 0; f < 4; f++) attw16[j][f] = aF16(256 * 256);

    float*     h1    = aF32((size_t)M_ * 128);
    float*     hf    = aF32((size_t)M_ * 256);
    float*     hb    = aF32((size_t)M_ * 256);
    float*     hb2   = aF32((size_t)M_ * 256);
    _Float16*  xln16 = aF16((size_t)M_ * 256);
    float*     xz    = aF32((size_t)M_ * 512);
    float*     xia   = aF32((size_t)M_ * 256);
    _Float16*  xia16 = aF16((size_t)M_ * 256);
    float*     dbc   = aF32((size_t)M_ * 48);
    float*     dtb   = aF32((size_t)M_ * 256);
    _Float16*  y16   = aF16((size_t)M_ * 256);
    _Float16*  cat16 = aF16((size_t)M_ * 512);
    _Float16*  enc16 = aF16((size_t)M_ * 256);
    float*     hh    = aF32((size_t)M_ * 256);
    float*     gate  = aF32((size_t)M_);
    float*     qb_   = aF32((size_t)M_ * 256);
    float*     kb_   = aF32((size_t)M_ * 256);
    float*     vb_   = aF32((size_t)M_ * 256);
    _Float16*  o16   = aF16((size_t)M_ * 256);
    float*     oproj = aF32((size_t)M_ * 256);
    float*     abuf  = aF32((size_t)M_ * 256);
    _Float16*  a16   = aF16((size_t)M_ * 256);

    float* out       = (float*)d_out;
    float* coarseOut = out;                  // [B,L,3]
    float* refinedOut= out + (size_t)M_ * 3;
    float* encOut    = out + (size_t)M_ * 6; // [B,L,256]

    auto cvt = [&](const float* s, _Float16* d, int n) {
        cvt_kernel<<<(n + 255) / 256, 256, 0, stream>>>(s, d, n);
    };
    auto gemm = [&](const _Float16* A, const _Float16* W, const float* bias,
                    const float* res, float* C, _Float16* C16,
                    int M, int N, int K, int act) {
        int tilesN = N / 16;
        int jobs   = (M / (16 * MT)) * ((tilesN + NT - 1) / NT);
        gemm_wmma_kernel<<<(jobs + 3) / 4, 128, 0, stream>>>(A, W, bias, res, C, C16,
                                                             M, N, K, act);
    };

    // ---- weight conversion (f32 -> f16) ----
    for (int dir = 0; dir < 2; dir++)
        for (int i = 0; i < 6; i++) {
            int li = dir * 6 + i;
            cvt(LYR(dir, i, 2), inproj16[li], 512 * 256);
            cvt(LYR(dir, i, 5), xproj16[li], 48 * 256);
            cvt(LYR(dir, i, 10), outproj16[li], 256 * 256);
        }
    cvt(P[145], fusion16, 256 * 512);
    cvt(P[146], cw1_16, 256 * 256);
    cvt(P[170], rw1_16, 256 * 256);
    for (int j = 0; j < 2; j++) {
        cvt(ATT(j, 0), attw16[j][0], 256 * 256);
        cvt(ATT(j, 2), attw16[j][1], 256 * 256);
        cvt(ATT(j, 4), attw16[j][2], 256 * 256);
        cvt(ATT(j, 6), attw16[j][3], 256 * 256);
    }

    // ---- stem + posenc (writes hf and reversed hb) ----
    stem1_kernel<<<(M_ * 128 + 255) / 256, 256, 0, stream>>>(
        P[0], P[1], P[2], P[3], P[4], P[5], P[6], h1);
    stem2_kernel<<<(M_ * 256 + 255) / 256, 256, 0, stream>>>(
        h1, P[7], P[8], P[9], P[10], P[11], P[12], hf, hb);

    // ---- mamba layers ----
    auto run_mamba = [&](float* strm, int dir, int i) {
        int li = dir * 6 + i;
        ln_kernel<<<M_ / 8, 256, 0, stream>>>(strm, nullptr, nullptr,
                                              LYR(dir, i, 0), LYR(dir, i, 1),
                                              nullptr, xln16);
        gemm(xln16, inproj16[li], nullptr, nullptr, xz, nullptr, M_, 512, 256, 0);
        dwconv_silu_kernel<<<(M_ * 256 + 255) / 256, 256, 0, stream>>>(
            xz, LYR(dir, i, 3), LYR(dir, i, 4), xia, xia16);
        gemm(xia16, xproj16[li], nullptr, nullptr, dbc, nullptr, M_, 48, 256, 0);
        dt_kernel<<<(M_ * 256 + 255) / 256, 256, 0, stream>>>(
            dbc, LYR(dir, i, 6), LYR(dir, i, 7), dtb);
        ssm_scan_kernel<<<B_, 256, 0, stream>>>(dtb, dbc, xia, xz,
                                                LYR(dir, i, 8), LYR(dir, i, 9), y16);
        gemm(y16, outproj16[li], nullptr, strm, strm, nullptr, M_, 256, 256, 0);
    };
    for (int i = 0; i < 6; i++) run_mamba(hf, 0, i);
    for (int i = 0; i < 6; i++) run_mamba(hb, 1, i);
    reverse_kernel<<<(M_ * 256 + 255) / 256, 256, 0, stream>>>(hb, hb2);

    // ---- fusion -> enc (into d_out) ----
    concat16_kernel<<<(M_ * 512 + 255) / 256, 256, 0, stream>>>(hf, hb2, cat16);
    gemm(cat16, fusion16, nullptr, nullptr, encOut, enc16, M_, 256, 512, 0);

    // ---- coarse head + gate ----
    gemm(enc16, cw1_16, P[147], nullptr, hh, nullptr, M_, 256, 256, 1);
    head2_kernel<<<(M_ + 255) / 256, 256, 0, stream>>>(hh, P[148], P[149],
                                                       coarseOut, gate);

    // ---- attention stream init: a = enc ----
    hipMemcpyAsync(abuf, encOut, (size_t)M_ * 256 * sizeof(float),
                   hipMemcpyDeviceToDevice, stream);
    hipMemcpyAsync(a16, enc16, (size_t)M_ * 256 * sizeof(_Float16),
                   hipMemcpyDeviceToDevice, stream);

    for (int j = 0; j < 2; j++) {
        gemm(a16, attw16[j][0], ATT(j, 1), nullptr, qb_, nullptr, M_, 256, 256, 0);
        gemm(a16, attw16[j][1], ATT(j, 3), nullptr, kb_, nullptr, M_, 256, 256, 0);
        gemm(a16, attw16[j][2], ATT(j, 5), nullptr, vb_, nullptr, M_, 256, 256, 0);
        band_attn_kernel<<<(B_ * NHEADS * L_ * 32) / 256, 256, 0, stream>>>(
            qb_, kb_, vb_, o16);
        gemm(o16, attw16[j][3], ATT(j, 7), nullptr, oproj, nullptr, M_, 256, 256, 0);
        // a = LN(a*(1-g) + oproj*g)  (in-place, writes f32 + f16)
        ln_kernel<<<M_ / 8, 256, 0, stream>>>(abuf, oproj, gate,
                                              ATT(j, 8), ATT(j, 9), abuf, a16);
    }

    // ---- refined head ----
    gemm(a16, rw1_16, P[171], nullptr, hh, nullptr, M_, 256, 256, 1);
    head2_kernel<<<(M_ + 255) / 256, 256, 0, stream>>>(hh, P[172], P[173],
                                                       refinedOut, nullptr);
}
